// NodeAttAggregator_73469710565691
// MI455X (gfx1250) — matrix operand order repaired and checked by reading, not compiled, not run on hardware
//
#include <hip/hip_runtime.h>
#include <hip/hip_bf16.h>

#define E_N     8192
#define IN_D    256
#define OUT_D   128

typedef __attribute__((ext_vector_type(16))) _Float16 v16h;
typedef __attribute__((ext_vector_type(8)))  _Float16 v8h;
typedef __attribute__((ext_vector_type(8)))  float    v8f;
typedef __attribute__((ext_vector_type(4)))  float    v4f;

union V16HU { v16h v; v8h h[2]; };

// Build a 16-element f16 fragment from two contiguous 16-byte chunks (b128 loads).
__device__ __forceinline__ v16h frag16(const _Float16* p0, const _Float16* p1) {
    V16HU u;
    u.h[0] = *(const v8h*)p0;
    u.h[1] = *(const v8h*)p1;
    return u.v;
}

// Load 8 contiguous f32 and convert to packed f16.
__device__ __forceinline__ v8h cvt8(const float* p) {
    v4f a = *(const v4f*)p;
    v4f b = *(const v4f*)(p + 4);
    v8h r;
    r[0]=(_Float16)a[0]; r[1]=(_Float16)a[1]; r[2]=(_Float16)a[2]; r[3]=(_Float16)a[3];
    r[4]=(_Float16)b[0]; r[5]=(_Float16)b[1]; r[6]=(_Float16)b[2]; r[7]=(_Float16)b[3];
    return r;
}

// XOR lane swizzle inside the wave (group-of-32 mode, and_mask=0x1f).
#define XOR_SWZ_F(x, m) \
    __int_as_float(__builtin_amdgcn_ds_swizzle(__float_as_int(x), (((m) << 10) | 0x1f)))

// ---------------------------------------------------------------------------
// Kernel 0: convert + transpose both projection weights to f16 [128][256]
// ---------------------------------------------------------------------------
__global__ __launch_bounds__(256)
void wtrans_kernel(const float* __restrict__ Wq, const float* __restrict__ Wk,
                   _Float16* __restrict__ WqT, _Float16* __restrict__ WkT) {
    int idx = blockIdx.x * blockDim.x + threadIdx.x;  // 2 * 128 * 256 total
    int sel = idx >> 15;
    int r   = idx & 32767;
    int n   = r >> 8;   // output column 0..127
    int k   = r & 255;  // input dim 0..255
    const float* W  = sel ? Wk  : Wq;
    _Float16*    WT = sel ? WkT : WqT;
    WT[(size_t)n * IN_D + k] = (_Float16)W[(size_t)k * OUT_D + n];
}

// ---------------------------------------------------------------------------
// Kernel 1: WMMA projection.  Block = 8 waves, covers 16 rows of X.
// Waves 0-3 -> Q columns, waves 4-7 -> K columns (32 cols per wave).
// Writes q_f16 [E][128], k_f16 [E][128], kT_f16 [128][E].
// ---------------------------------------------------------------------------
__global__ __launch_bounds__(256)
void proj_kernel(const float* __restrict__ X,
                 const _Float16* __restrict__ WqT, const _Float16* __restrict__ WkT,
                 _Float16* __restrict__ qh, _Float16* __restrict__ kh,
                 _Float16* __restrict__ kT) {
    const int lane = threadIdx.x & 31;
    const int wave = threadIdx.x >> 5;
    const int n15  = lane & 15;
    const int h    = lane >> 4;
    const int mBase = blockIdx.x * 16;
    const bool isK  = wave >= 4;
    const int colBase = (wave & 3) * 32;
    const _Float16* WT = isK ? WkT : WqT;

    v8f c0 = {}; v8f c1 = {};
    const float* xrow = X + (size_t)(mBase + n15) * IN_D;
    #pragma unroll
    for (int kf = 0; kf < IN_D / 32; ++kf) {
        const int kBase = kf * 32;
        v16h a;
        {   // A fragment of X (f32 -> f16 on the fly): chunks at k+16h and k+16h+8
            const float* p0 = xrow + kBase + 16 * h;
            V16HU u; u.h[0] = cvt8(p0); u.h[1] = cvt8(p0 + 8); a = u.v;
        }
        const _Float16* b0p = WT + (size_t)(colBase + n15) * IN_D + kBase + 16 * h;
        const _Float16* b1p = WT + (size_t)(colBase + 16 + n15) * IN_D + kBase + 16 * h;
        v16h b0 = frag16(b0p, b0p + 8);
        v16h b1 = frag16(b1p, b1p + 8);
        c0 = __builtin_amdgcn_wmma_f32_16x16x32_f16(false, a, false, b0, (short)0, c0, false, false);
        c1 = __builtin_amdgcn_wmma_f32_16x16x32_f16(false, a, false, b1, (short)0, c1, false, false);
    }
    _Float16* dst = isK ? kh : qh;
    #pragma unroll
    for (int t = 0; t < 2; ++t) {
        v8f c = t ? c1 : c0;
        const int col = colBase + t * 16 + n15;
        #pragma unroll
        for (int v = 0; v < 8; ++v)
            dst[(size_t)(mBase + v + 8 * h) * OUT_D + col] = (_Float16)c[v];
        if (isK) {  // C-layout rows (v contiguous) are contiguous in kT -> one b128 store
            v8h pk;
            #pragma unroll
            for (int v = 0; v < 8; ++v) pk[v] = (_Float16)c[v];
            *(v8h*)(kT + (size_t)col * E_N + mBase + 8 * h) = pk;
        }
    }
}

// ---------------------------------------------------------------------------
// Kernel 2: flash attention.  Each wave owns a 16-row query tile and streams
// 32-key blocks: S = Q@K^T (WMMA), mask+scale, online softmax, O += P@K (WMMA).
// ---------------------------------------------------------------------------
#define ATT_WAVES 4
__global__ __launch_bounds__(ATT_WAVES * 32)
void flash_attn_kernel(const _Float16* __restrict__ qh, const _Float16* __restrict__ kh,
                       const _Float16* __restrict__ kT, const int* __restrict__ mask,
                       float* __restrict__ out) {
    const int lane  = threadIdx.x & 31;
    const int wave  = threadIdx.x >> 5;
    const int n15   = lane & 15;
    const int h     = lane >> 4;
    const int mBase = (blockIdx.x * ATT_WAVES + wave) * 16;

    __shared__ __align__(16) _Float16 pLds[ATT_WAVES * 16 * 32];
    _Float16* pw = pLds + wave * 16 * 32;

    // Resident Q fragments for this tile: D=128 -> 4 fragments of K=32.
    v16h qf[4];
    {
        const _Float16* qrow = qh + (size_t)(mBase + n15) * OUT_D;
        #pragma unroll
        for (int f = 0; f < 4; ++f) {
            const _Float16* p0 = qrow + f * 32 + 16 * h;
            qf[f] = frag16(p0, p0 + 8);
        }
    }

    const float scale   = 0.08838834764831845f;  // 1/sqrt(128)
    const float NEG_BIG = -1.0e30f;

    float run_m[8], run_l[8];
    v8f zero = {};
    v8f o[8];
    #pragma unroll
    for (int v = 0; v < 8; ++v) { run_m[v] = NEG_BIG; run_l[v] = 0.0f; }
    #pragma unroll
    for (int t = 0; t < 8; ++t) o[t] = zero;

    const int NKB = E_N / 32;
    for (int nb = 0; nb < NKB; ++nb) {
        const int nBase = nb * 32;
        if (nb + 1 < NKB)  // prefetch next mask tile (global_prefetch_b8)
            __builtin_prefetch(mask + (size_t)(mBase + n15) * E_N + nBase + 32 + h * 16, 0, 1);

        // --- S = Q @ K^T : two 16x16 n-tiles, 8 WMMAs ---
        v8f s0 = {}; v8f s1 = {};
        #pragma unroll
        for (int f = 0; f < 4; ++f) {
            const _Float16* b0p = kh + (size_t)(nBase + n15) * OUT_D + f * 32 + 16 * h;
            const _Float16* b1p = kh + (size_t)(nBase + 16 + n15) * OUT_D + f * 32 + 16 * h;
            v16h b0 = frag16(b0p, b0p + 8);
            v16h b1 = frag16(b1p, b1p + 8);
            s0 = __builtin_amdgcn_wmma_f32_16x16x32_f16(false, qf[f], false, b0, (short)0, s0, false, false);
            s1 = __builtin_amdgcn_wmma_f32_16x16x32_f16(false, qf[f], false, b1, (short)0, s1, false, false);
        }

        // --- scale + mask (C layout: row m = v + 8h, col = tile*16 + n15) ---
        float sc[2][8];
        #pragma unroll
        for (int v = 0; v < 8; ++v) {
            const size_t rowoff = (size_t)(mBase + v + 8 * h) * E_N + nBase;
            int m0 = mask[rowoff + n15];
            int m1 = mask[rowoff + 16 + n15];
            sc[0][v] = m0 ? s0[v] * scale : NEG_BIG;
            sc[1][v] = m1 ? s1[v] * scale : NEG_BIG;
        }

        // --- online softmax: row reductions across the 16 lanes of each half ---
        float nm[8], rs[8];
        #pragma unroll
        for (int v = 0; v < 8; ++v) {
            float t0 = fmaxf(sc[0][v], sc[1][v]);
            t0 = fmaxf(t0, XOR_SWZ_F(t0, 1));
            t0 = fmaxf(t0, XOR_SWZ_F(t0, 2));
            t0 = fmaxf(t0, XOR_SWZ_F(t0, 4));
            t0 = fmaxf(t0, XOR_SWZ_F(t0, 8));
            nm[v] = fmaxf(t0, run_m[v]);
        }
        #pragma unroll
        for (int v = 0; v < 8; ++v) {
            float corr = __expf(run_m[v] - nm[v]);
            float p0 = __expf(sc[0][v] - nm[v]);
            float p1 = __expf(sc[1][v] - nm[v]);
            float ps = p0 + p1;
            ps += XOR_SWZ_F(ps, 1);
            ps += XOR_SWZ_F(ps, 2);
            ps += XOR_SWZ_F(ps, 4);
            ps += XOR_SWZ_F(ps, 8);
            run_l[v] = run_l[v] * corr + ps;
            run_m[v] = nm[v];
            rs[v] = corr;
            sc[0][v] = p0; sc[1][v] = p1;
        }
        #pragma unroll
        for (int t = 0; t < 8; ++t)
            #pragma unroll
            for (int v = 0; v < 8; ++v) o[t][v] *= rs[v];

        // --- P: C layout -> LDS -> A layout ---
        #pragma unroll
        for (int t2 = 0; t2 < 2; ++t2)
            #pragma unroll
            for (int v = 0; v < 8; ++v)
                pw[(v + 8 * h) * 32 + t2 * 16 + n15] = (_Float16)sc[t2][v];
        __asm__ volatile("s_wait_dscnt 0" ::: "memory");
        v16h pa = frag16(pw + n15 * 32 + 16 * h, pw + n15 * 32 + 8 + 16 * h);

        // --- O += P @ K : B fragments from kT (contiguous along key index), 8 WMMAs ---
        #pragma unroll
        for (int t = 0; t < 8; ++t) {
            const _Float16* bp = kT + (size_t)(t * 16 + n15) * E_N + nBase + 16 * h;
            v16h b = frag16(bp, bp + 8);
            o[t] = __builtin_amdgcn_wmma_f32_16x16x32_f16(false, pa, false, b, (short)0, o[t], false, false);
        }
    }

    // --- normalize + store f32 output ---
    float inv[8];
    #pragma unroll
    for (int v = 0; v < 8; ++v) inv[v] = 1.0f / run_l[v];
    #pragma unroll
    for (int t = 0; t < 8; ++t)
        #pragma unroll
        for (int v = 0; v < 8; ++v)
            out[(size_t)(mBase + v + 8 * h) * OUT_D + t * 16 + n15] = o[t][v] * inv[v];
}

// ---------------------------------------------------------------------------
extern "C" void kernel_launch(void* const* d_in, const int* in_sizes, int n_in,
                              void* d_out, int out_size, void* d_ws, size_t ws_size,
                              hipStream_t stream) {
    const float* X    = (const float*)d_in[0];   // [8192, 256]
    const float* Wq   = (const float*)d_in[1];   // [256, 128]
    const float* Wk   = (const float*)d_in[2];   // [256, 128]
    const int*   mask = (const int*)d_in[3];     // [8192, 8192]
    float* out = (float*)d_out;                  // [8192, 128]

    char* ws = (char*)d_ws;
    _Float16* WqT = (_Float16*)(ws);                              //  64 KB
    _Float16* WkT = (_Float16*)(ws + 65536);                      //  64 KB
    _Float16* qh  = (_Float16*)(ws + 131072);                     //   2 MB
    _Float16* kh  = (_Float16*)(ws + 131072 + 2097152);           //   2 MB
    _Float16* kT  = (_Float16*)(ws + 131072 + 2 * 2097152);       //   2 MB

    hipLaunchKernelGGL(wtrans_kernel, dim3(256), dim3(256), 0, stream, Wq, Wk, WqT, WkT);
    hipLaunchKernelGGL(proj_kernel, dim3(E_N / 16), dim3(256), 0, stream,
                       X, WqT, WkT, qh, kh, kT);
    hipLaunchKernelGGL(flash_attn_kernel, dim3(E_N / (16 * ATT_WAVES)), dim3(ATT_WAVES * 32), 0, stream,
                       qh, kh, kT, mask, out);
}
